// DoRAAdapter_18373870092444
// MI455X (gfx1250) — matrix operand order repaired and checked by reading, not compiled, not run on hardware
//
#include <hip/hip_runtime.h>
#include <hip/hip_bf16.h>
#include <stdint.h>

// ---------------- problem dims (fixed by the reference) ----------------
#define ALPHA_F 16.0f
#define MTOT    8192      // B*S = 4*2048 tokens
#define IN_     4096
#define OUT_    4096
#define R_      16

// ---------------- WMMA vector types (gfx1250, wave32) ------------------
typedef __attribute__((ext_vector_type(16))) __bf16 v16bf;
typedef __attribute__((ext_vector_type(8)))  float  v8f;

struct BF16x16Bits { uint4 lo, hi; };   // 32 bytes == v16bf

// round-to-nearest-even fp32 -> bf16 (bit trick; inputs are finite normals)
__device__ __forceinline__ unsigned int f2bf(float f) {
  union { float f; unsigned int u; } v; v.f = f;
  unsigned int u = v.u;
  u += 0x7FFFu + ((u >> 16) & 1u);
  return u >> 16;
}

// ---- CDNA5 async global->LDS copy (ASYNCcnt-tracked, ISA ch.8) --------
__device__ __forceinline__ void async_b128(unsigned lds_addr, const void* gaddr) {
  asm volatile("global_load_async_to_lds_b128 %0, %1, off"
               :: "v"(lds_addr), "v"(gaddr) : "memory");
}

#if __has_builtin(__builtin_amdgcn_s_wait_asynccnt)
#define WAIT_ASYNC(n) __builtin_amdgcn_s_wait_asynccnt(n)
#else
#define WAIT_ASYNC(n) asm volatile("s_wait_asynccnt %0" :: "i"(n) : "memory")
#endif

// =======================================================================
// Kernel 1: deterministic partial column sum-of-squares of
//   numerator[o,i] = weight[o,i] + ALPHA * (dora_B[o,:] . dora_A[:,i])
// grid = (IN/256, 8); block handles 256 columns over OUT/8 rows.
// =======================================================================
__global__ __launch_bounds__(256)
void colnorm_partial(const float* __restrict__ weight,
                     const float* __restrict__ dA,   // [16][IN]
                     const float* __restrict__ dB,   // [OUT][16]
                     float* __restrict__ partials)   // [8][IN]
{
  __shared__ float sB[64 * 16];
  const int tid = threadIdx.x;
  const int i   = blockIdx.x * 256 + tid;            // my column
  const int o0b = blockIdx.y * (OUT_ / 8);

  float a[R_];
  #pragma unroll
  for (int r = 0; r < R_; ++r) a[r] = dA[r * IN_ + i];

  float acc = 0.f;
  for (int o0 = o0b; o0 < o0b + OUT_ / 8; o0 += 64) {
    __syncthreads();
    ((float4*)sB)[tid] = ((const float4*)(dB + (size_t)o0 * R_))[tid]; // 1024 floats
    __syncthreads();
    #pragma unroll 4
    for (int oo = 0; oo < 64; ++oo) {
      float dot = 0.f;
      #pragma unroll
      for (int r = 0; r < R_; ++r) dot = fmaf(sB[oo * R_ + r], a[r], dot);
      float num = weight[(size_t)(o0 + oo) * IN_ + i] + ALPHA_F * dot;
      acc = fmaf(num, num, acc);
    }
  }
  partials[(size_t)blockIdx.y * IN_ + i] = acc;       // no atomics: deterministic
}

// =======================================================================
// Kernel 2: W'bf16[o,i] = bf16( m[i]*rsqrt(colsumsq[i]) * numerator[o,i] )
// one thread -> 8 consecutive i of one row o; 16B stores.
// =======================================================================
__global__ __launch_bounds__(256)
void prep_w(const float* __restrict__ weight,
            const float* __restrict__ dA,
            const float* __restrict__ dB,
            const float* __restrict__ m,
            const float* __restrict__ partials,       // [8][IN]
            unsigned short* __restrict__ wbf)         // [OUT][IN] bf16
{
  const int gid = blockIdx.x * 256 + threadIdx.x;
  const int o   = gid >> 9;                           // 512 groups of 8 per row
  const int i0  = (gid & 511) << 3;

  float b[R_];
  #pragma unroll
  for (int r = 0; r < R_; ++r) b[r] = dB[(size_t)o * R_ + r];   // uniform per block

  float d[8];
  #pragma unroll
  for (int j = 0; j < 8; ++j) d[j] = 0.f;
  #pragma unroll
  for (int r = 0; r < R_; ++r) {
    float4 a0 = *(const float4*)(dA + (size_t)r * IN_ + i0);
    float4 a1 = *(const float4*)(dA + (size_t)r * IN_ + i0 + 4);
    d[0] = fmaf(b[r], a0.x, d[0]); d[1] = fmaf(b[r], a0.y, d[1]);
    d[2] = fmaf(b[r], a0.z, d[2]); d[3] = fmaf(b[r], a0.w, d[3]);
    d[4] = fmaf(b[r], a1.x, d[4]); d[5] = fmaf(b[r], a1.y, d[5]);
    d[6] = fmaf(b[r], a1.z, d[6]); d[7] = fmaf(b[r], a1.w, d[7]);
  }

  float ss[8];
  #pragma unroll
  for (int j = 0; j < 8; ++j) ss[j] = 0.f;
  #pragma unroll
  for (int p = 0; p < 8; ++p) {
    float4 s0 = *(const float4*)(partials + (size_t)p * IN_ + i0);
    float4 s1 = *(const float4*)(partials + (size_t)p * IN_ + i0 + 4);
    ss[0] += s0.x; ss[1] += s0.y; ss[2] += s0.z; ss[3] += s0.w;
    ss[4] += s1.x; ss[5] += s1.y; ss[6] += s1.z; ss[7] += s1.w;
  }

  float4 w0 = *(const float4*)(weight + (size_t)o * IN_ + i0);
  float4 w1 = *(const float4*)(weight + (size_t)o * IN_ + i0 + 4);
  float4 m0 = *(const float4*)(m + i0);
  float4 m1 = *(const float4*)(m + i0 + 4);
  float num[8] = { w0.x + ALPHA_F*d[0], w0.y + ALPHA_F*d[1],
                   w0.z + ALPHA_F*d[2], w0.w + ALPHA_F*d[3],
                   w1.x + ALPHA_F*d[4], w1.y + ALPHA_F*d[5],
                   w1.z + ALPHA_F*d[6], w1.w + ALPHA_F*d[7] };
  float mm[8]  = { m0.x, m0.y, m0.z, m0.w, m1.x, m1.y, m1.z, m1.w };

  unsigned int h[8];
  #pragma unroll
  for (int j = 0; j < 8; ++j)
    h[j] = f2bf(mm[j] * __frsqrt_rn(ss[j]) * num[j]);

  uint4 outv;
  outv.x = h[0] | (h[1] << 16);
  outv.y = h[2] | (h[3] << 16);
  outv.z = h[4] | (h[5] << 16);
  outv.w = h[6] | (h[7] << 16);
  *(uint4*)(wbf + (size_t)o * IN_ + i0) = outv;
}

// =======================================================================
// Kernel 3: x fp32 -> bf16 (8 elems / thread, 16B stores)
// =======================================================================
__global__ __launch_bounds__(256)
void conv_x(const float* __restrict__ x, unsigned short* __restrict__ xb)
{
  const size_t i = ((size_t)blockIdx.x * 256 + threadIdx.x) * 8;
  float4 a = *(const float4*)(x + i);
  float4 b = *(const float4*)(x + i + 4);
  uint4 o;
  o.x = f2bf(a.x) | (f2bf(a.y) << 16);
  o.y = f2bf(a.z) | (f2bf(a.w) << 16);
  o.z = f2bf(b.x) | (f2bf(b.y) << 16);
  o.w = f2bf(b.z) | (f2bf(b.w) << 16);
  *(uint4*)(xb + i) = o;
}

// =======================================================================
// Kernel 4: C[MTOT,OUT] = Xbf16 * W'bf16^T + bias   (fp32 accumulate)
// 256 threads = 8 wave32 (2 M-strips x 4 N-strips), wave tile 64x64
// (16 fp32 WMMA accumulators). Block tile 128x256, K staged 32 per
// buffer, double-buffered via GLOBAL_LOAD_ASYNC_TO_LDS_B128 + ASYNCcnt.
// All 8 fragments are pre-loaded per stage so the 16 ds_load_b128 clause
// together and the 16 WMMAs issue back-to-back without dscnt stalls.
// =======================================================================
#define MT  128
#define NTB 256
#define KT  32
#define PAe 40
#define PBe 40

__global__ __launch_bounds__(256)
void gemm_bf16(const unsigned short* __restrict__ xb,   // [MTOT][IN] bf16
               const unsigned short* __restrict__ wb,   // [OUT][IN]  bf16
               const float* __restrict__ bias,
               float* __restrict__ out)                 // [MTOT][OUT] f32
{
  __shared__ __align__(16) unsigned short As[2][MT  * PAe];  // 2 x 10 KB
  __shared__ __align__(16) unsigned short Bs[2][NTB * PBe];  // 2 x 20 KB

  const int tid  = threadIdx.x;
  const int lane = tid & 31;
  const int wave = tid >> 5;           // 0..7
  const int wm   = wave >> 2;          // 0..1  (64-row strip)
  const int wn   = wave & 3;           // 0..3  (64-col strip)
  const int m0   = blockIdx.y * MT;
  const int n0   = blockIdx.x * NTB;
  const int laneM = lane & 15;
  const int laneH = lane >> 4;

  // ---- async staging roles -------------------------------------------
  // A tile 128x32 bf16: thread -> (row = tid/2, 16-elem half), 2x b128
  // B tile 256x32 bf16: thread -> row tid, 4x b128
  const int srowA = tid >> 1;
  const int khA   = (tid & 1) * 16;
  const unsigned aOff0 = (unsigned)(uintptr_t)(&As[0][0]) +
                         (unsigned)(srowA * PAe + khA) * 2u;
  const unsigned bOff0 = (unsigned)(uintptr_t)(&Bs[0][0]) +
                         (unsigned)(tid * PBe) * 2u;
  const unsigned aStride = MT  * PAe * 2u;   // bytes between buffers
  const unsigned bStride = NTB * PBe * 2u;
  const unsigned short* gA = xb + (size_t)(m0 + srowA) * IN_ + khA;
  const unsigned short* gB = wb + (size_t)(n0 + tid)   * IN_;

  v8f acc[4][4];
  #pragma unroll
  for (int mi = 0; mi < 4; ++mi)
    #pragma unroll
    for (int ni = 0; ni < 4; ++ni) acc[mi][ni] = (v8f){0,0,0,0,0,0,0,0};

  // prologue: fill buffer 0 with k = 0  (6 async ops / thread / stage)
  async_b128(aOff0,      gA);
  async_b128(aOff0 + 16, gA + 8);
  async_b128(bOff0,      gB);
  async_b128(bOff0 + 16, gB + 8);
  async_b128(bOff0 + 32, gB + 16);
  async_b128(bOff0 + 48, gB + 24);

  int cur = 0;
  for (int k0 = 0; k0 < IN_; k0 += KT) {
    const int nxt = cur ^ 1;
    if (k0 + KT < IN_) {
      const unsigned short* gAn = gA + k0 + KT;
      const unsigned short* gBn = gB + k0 + KT;
      const unsigned aO = aOff0 + (unsigned)nxt * aStride;
      const unsigned bO = bOff0 + (unsigned)nxt * bStride;
      async_b128(aO,      gAn);
      async_b128(aO + 16, gAn + 8);
      async_b128(bO,      gBn);
      async_b128(bO + 16, gBn + 8);
      async_b128(bO + 32, gBn + 16);
      async_b128(bO + 48, gBn + 24);
      WAIT_ASYNC(6);          // in-order: the 6 older (cur-buffer) are done
    } else {
      WAIT_ASYNC(0);          // last stage: drain everything
    }
    __syncthreads();          // all waves' cur-buffer data visible

    // ---- compute one 32-deep K slab ----------------------------------
    // ISA fragment layouts (wave32):
    //  A 16x32 bf16: lane holds M=lane%16; K chunks (lane/16)*8 and +16
    //  B 32x16 bf16: lane holds N=lane%16; K chunks (lane/16)*8 and +16
    {
      const unsigned short* Ac = As[cur];
      const unsigned short* Bc = Bs[cur];
      const int kA = laneH * 8;
      const int kB = laneH * 8;

      v16bf af[4], bfr[4];
      #pragma unroll
      for (int mi = 0; mi < 4; ++mi) {
        const int row = wm * 64 + mi * 16 + laneM;
        BF16x16Bits t;
        t.lo = *(const uint4*)(Ac + row * PAe + kA);
        t.hi = *(const uint4*)(Ac + row * PAe + kA + 16);
        af[mi] = __builtin_bit_cast(v16bf, t);
      }
      #pragma unroll
      for (int ni = 0; ni < 4; ++ni) {
        const int row = wn * 64 + ni * 16 + laneM;   // B column n == W' row n
        BF16x16Bits t;
        t.lo = *(const uint4*)(Bc + row * PBe + kB);
        t.hi = *(const uint4*)(Bc + row * PBe + kB + 16);
        bfr[ni] = __builtin_bit_cast(v16bf, t);
      }
      // 16 independent WMMAs, no ds waits interleaved
      #pragma unroll
      for (int ni = 0; ni < 4; ++ni)
        #pragma unroll
        for (int mi = 0; mi < 4; ++mi)
          acc[mi][ni] = __builtin_amdgcn_wmma_f32_16x16x32_bf16(
              false, af[mi], false, bfr[ni], (short)0, acc[mi][ni],
              false, false);
    }
    __syncthreads();          // all waves done reading before refill
    cur = nxt;
  }

  // epilogue: C layout — VGPR r: M = r + 8*(lane>>4); N = lane&15
  #pragma unroll
  for (int mi = 0; mi < 4; ++mi)
    #pragma unroll
    for (int ni = 0; ni < 4; ++ni) {
      const int n  = n0 + wn * 64 + ni * 16 + laneM;
      const float bv = bias[n];
      #pragma unroll
      for (int r = 0; r < 8; ++r) {
        const int mrow = m0 + wm * 64 + mi * 16 + r + 8 * laneH;
        out[(size_t)mrow * OUT_ + n] = acc[mi][ni][r] + bv;
      }
    }
}

// =======================================================================
extern "C" void kernel_launch(void* const* d_in, const int* in_sizes, int n_in,
                              void* d_out, int out_size, void* d_ws, size_t ws_size,
                              hipStream_t stream) {
  const float* x    = (const float*)d_in[0];   // [4,2048,4096]
  const float* w    = (const float*)d_in[1];   // [4096,4096]
  const float* bias = (const float*)d_in[2];   // [4096]
  const float* m    = (const float*)d_in[3];   // [1,4096]
  const float* dA   = (const float*)d_in[4];   // [16,4096]
  const float* dB   = (const float*)d_in[5];   // [4096,16]
  float* out = (float*)d_out;

  char* ws = (char*)d_ws;
  float*          partials = (float*)ws;                          // 8*IN f32 (128 KB)
  unsigned short* wbf = (unsigned short*)(ws + (size_t)8 * IN_ * sizeof(float));
  unsigned short* xbf = wbf + (size_t)OUT_ * IN_;                 // +32 MB, then 64 MB

  colnorm_partial<<<dim3(IN_ / 256, 8), 256, 0, stream>>>(w, dA, dB, partials);
  prep_w<<<(OUT_ * (IN_ / 8)) / 256, 256, 0, stream>>>(w, dA, dB, m, partials, wbf);
  conv_x<<<(MTOT * IN_ / 8) / 256, 256, 0, stream>>>(x, xbf);
  gemm_bf16<<<dim3(OUT_ / NTB, MTOT / MT), 256, 0, stream>>>(xbf, wbf, bias, out);
}
